// QuantumTabularModel_5652176962206
// MI455X (gfx1250) — compile-verified, matrix operand order checked by reading, not run on hardware
//
#include <hip/hip_runtime.h>
#include <math.h>

#define BATCH   8192
#define IN_DIM  128
#define QDIM    32
#define NQ      8
#define OUT_DIM 9
#define NAMP    256          // 2^NQ amplitudes

typedef __attribute__((ext_vector_type(16))) _Float16 v16h;
typedef __attribute__((ext_vector_type(8)))  float    v8f;
typedef int v4i_ __attribute__((ext_vector_type(4)));

#if defined(__AMDGCN__) && __has_builtin(__builtin_amdgcn_global_load_async_to_lds_b128)
#define HAVE_ASYNC_LDS 1
#endif

// one 16B chunk: global -> LDS (async on CDNA5, sync fallback otherwise)
__device__ __forceinline__ void copy16_async(const _Float16* g, _Float16* l)
{
#ifdef HAVE_ASYNC_LDS
    __builtin_amdgcn_global_load_async_to_lds_b128(
        (__attribute__((address_space(1))) v4i_*)(void*)(g),
        (__attribute__((address_space(3))) v4i_*)(void*)(l), 0, 0);
#else
    *(uint4*)l = *(const uint4*)g;
#endif
}

__device__ __forceinline__ void wait_async_barrier()
{
#ifdef HAVE_ASYNC_LDS
#if __has_builtin(__builtin_amdgcn_s_wait_asynccnt)
    __builtin_amdgcn_s_wait_asynccnt(0);
#endif
#endif
    __syncthreads();
}

// ---------------- inter-kernel buffers (fully rewritten every call) ----------------
__device__ float    g_feats[BATCH * NQ];          // 256 KB
__device__ _Float16 g_PSI[BATCH * 2 * NAMP];      // 8 MB: [s][k] k<256:Re, k>=256:Im
__device__ _Float16 g_UR[NAMP * NAMP];            // [i][j] = Re U[j][i]  (i=input amp)
__device__ _Float16 g_UI[NAMP * NAMP];            // [i][j] = Im U[j][i]
__device__ _Float16 g_BWR[16 * 16 * 32 * 16];     // B_R in WMMA fragment layout
__device__ _Float16 g_BWI[16 * 16 * 32 * 16];     // B_I in WMMA fragment layout
__device__ float    g_Z[BATCH * NQ];              // <Z_q> per sample

// =====================================================================
// Kernel 1: feats = tanh(relu(x @ w1^T + b1) @ w2^T + b2)  via WMMA f16
// =====================================================================
__global__ __launch_bounds__(256) void feats_kernel(
    const float* __restrict__ x,  const float* __restrict__ w1,
    const float* __restrict__ b1, const float* __restrict__ w2,
    const float* __restrict__ b2)
{
    __shared__ _Float16 ldsH[8][16 * 32];

    const int lane  = threadIdx.x & 31;
    const int wave  = threadIdx.x >> 5;
    const int row0  = (blockIdx.x * 8 + wave) * 16;
    const int mrow  = lane & 15;
    const int kbase = (lane >> 4) * 8;
    const int m0    = (lane >> 4) * 8;

    v16h A[4];
#pragma unroll
    for (int kc = 0; kc < 4; ++kc) {
        const float* xp = x + (size_t)(row0 + mrow) * IN_DIM + kc * 32 + kbase;
#pragma unroll
        for (int j = 0; j < 8; ++j) {
            A[kc][j]     = (_Float16)xp[j];
            A[kc][8 + j] = (_Float16)xp[16 + j];
        }
    }

#pragma unroll
    for (int nt = 0; nt < 2; ++nt) {
        v8f acc = {};
#pragma unroll
        for (int kc = 0; kc < 4; ++kc) {
            v16h B;
            const float* wp = w1 + (size_t)(nt * 16 + mrow) * IN_DIM + kc * 32 + kbase;
#pragma unroll
            for (int j = 0; j < 8; ++j) {
                B[j]     = (_Float16)wp[j];
                B[8 + j] = (_Float16)wp[16 + j];
            }
            acc = __builtin_amdgcn_wmma_f32_16x16x32_f16(
                      false, A[kc], false, B, (short)0, acc, false, false);
        }
        const float bias = b1[nt * 16 + mrow];
#pragma unroll
        for (int r = 0; r < 8; ++r) {
            float v = acc[r] + bias;
            ldsH[wave][(m0 + r) * 32 + nt * 16 + mrow] = (_Float16)(v > 0.f ? v : 0.f);
        }
    }
    __syncthreads();

    v16h A2, B2;
    const _Float16* hp = &ldsH[wave][mrow * 32];
#pragma unroll
    for (int j = 0; j < 8; ++j) {
        A2[j]     = hp[kbase + j];
        A2[8 + j] = hp[kbase + 16 + j];
    }
    if (mrow < NQ) {
        const float* wp = w2 + (size_t)mrow * QDIM;
#pragma unroll
        for (int j = 0; j < 8; ++j) {
            B2[j]     = (_Float16)wp[kbase + j];
            B2[8 + j] = (_Float16)wp[kbase + 16 + j];
        }
    } else {
#pragma unroll
        for (int j = 0; j < 16; ++j) B2[j] = (_Float16)0.f;
    }
    v8f acc2 = {};
    acc2 = __builtin_amdgcn_wmma_f32_16x16x32_f16(
               false, A2, false, B2, (short)0, acc2, false, false);

    if (mrow < NQ) {
        const float bias = b2[mrow];
#pragma unroll
        for (int r = 0; r < 8; ++r)
            g_feats[(size_t)(row0 + m0 + r) * NQ + mrow] = tanhf(acc2[r] + bias);
    }
}

// =====================================================================
// Statevector gate machinery: amp index = (r<<5)|lane, qubit q -> bit 7-q
// =====================================================================
__device__ __forceinline__ void cgate(float re[8], float im[8], int lane,
    int pt, int pc,
    float m00r, float m00i, float m01r, float m01i,
    float m10r, float m10i, float m11r, float m11i)
{
    if (pt >= 5) {
        const int rm = 1 << (pt - 5);
#pragma unroll
        for (int r = 0; r < 8; ++r) {
            if (r & rm) continue;
            const int r1 = r | rm;
            bool apply = true;
            if (pc >= 0)
                apply = (pc >= 5) ? (((r >> (pc - 5)) & 1) != 0)
                                  : (((lane >> pc) & 1) != 0);
            float a0r = re[r],  a0i = im[r];
            float a1r = re[r1], a1i = im[r1];
            float n0r = m00r*a0r - m00i*a0i + m01r*a1r - m01i*a1i;
            float n0i = m00r*a0i + m00i*a0r + m01r*a1i + m01i*a1r;
            float n1r = m10r*a0r - m10i*a0i + m11r*a1r - m11i*a1i;
            float n1i = m10r*a0i + m10i*a0r + m11r*a1i + m11i*a1r;
            re[r]  = apply ? n0r : a0r;  im[r]  = apply ? n0i : a0i;
            re[r1] = apply ? n1r : a1r;  im[r1] = apply ? n1i : a1i;
        }
    } else {
        const int  lm = 1 << pt;
        const bool hi = (lane & lm) != 0;
#pragma unroll
        for (int r = 0; r < 8; ++r) {
            float mr = re[r], mi = im[r];
            float pr = __shfl_xor(mr, lm, 32);
            float pi = __shfl_xor(mi, lm, 32);
            float a0r = hi ? pr : mr, a0i = hi ? pi : mi;
            float a1r = hi ? mr : pr, a1i = hi ? mi : pi;
            float n0r = m00r*a0r - m00i*a0i + m01r*a1r - m01i*a1i;
            float n0i = m00r*a0i + m00i*a0r + m01r*a1i + m01i*a1r;
            float n1r = m10r*a0r - m10i*a0i + m11r*a1r - m11i*a1i;
            float n1i = m10r*a0i + m10i*a0r + m11r*a1i + m11i*a1r;
            float rr = hi ? n1r : n0r, ri = hi ? n1i : n0i;
            bool apply = true;
            if (pc >= 0)
                apply = (pc >= 5) ? (((r >> (pc - 5)) & 1) != 0)
                                  : (((lane >> pc) & 1) != 0);
            re[r] = apply ? rr : mr;
            im[r] = apply ? ri : mi;
        }
    }
}

__device__ __forceinline__ void g_ry(float re[8], float im[8], int lane, int q, float t) {
    float s, c; __sincosf(0.5f * t, &s, &c);
    cgate(re, im, lane, 7 - q, -1, c, 0.f, -s, 0.f, s, 0.f, c, 0.f);
}
__device__ __forceinline__ void g_rz(float re[8], float im[8], int lane, int q, float t) {
    float s, c; __sincosf(0.5f * t, &s, &c);
    cgate(re, im, lane, 7 - q, -1, c, -s, 0.f, 0.f, 0.f, 0.f, c, s);
}
__device__ __forceinline__ void g_rx(float re[8], float im[8], int lane, int q, float t) {
    float s, c; __sincosf(0.5f * t, &s, &c);
    cgate(re, im, lane, 7 - q, -1, c, 0.f, 0.f, -s, 0.f, -s, c, 0.f);
}
__device__ __forceinline__ void g_cnot(float re[8], float im[8], int lane, int c, int t) {
    cgate(re, im, lane, 7 - t, 7 - c, 0.f, 0.f, 1.f, 0.f, 1.f, 0.f, 0.f, 0.f);
}
__device__ __forceinline__ void g_cry(float re[8], float im[8], int lane, int c, int t, float th) {
    float s, cc; __sincosf(0.5f * th, &s, &cc);
    cgate(re, im, lane, 7 - t, 7 - c, cc, 0.f, -s, 0.f, s, 0.f, cc, 0.f);
}

// =====================================================================
// Kernel 2: feature-map circuit per sample (one wave each) -> g_PSI (f16)
// =====================================================================
__global__ __launch_bounds__(256) void fm_kernel(const float* __restrict__ fm_w)
{
    const int lane = threadIdx.x & 31;
    const int wave = threadIdx.x >> 5;
    const int s    = blockIdx.x * 8 + wave;

    float f[NQ];
#pragma unroll
    for (int i = 0; i < NQ; ++i) f[i] = g_feats[(size_t)s * NQ + i];

    float re[8], im[8];
#pragma unroll
    for (int r = 0; r < 8; ++r) { re[r] = 0.f; im[r] = 0.f; }
    re[0] = (lane == 0) ? 1.f : 0.f;

    for (int l = 0; l < 2; ++l) {
        const float* fw = fm_w + l * (NQ * 2);
#pragma unroll
        for (int i = 0; i < NQ; ++i) {
            g_ry(re, im, lane, i, f[i] * fw[2 * i + 0]);
            g_rz(re, im, lane, i, f[i] * fw[2 * i + 1]);
        }
#pragma unroll
        for (int i = 0; i < NQ - 1; ++i) g_cnot(re, im, lane, i, i + 1);
#pragma unroll
        for (int i = 0; i < NQ; ++i) g_ry(re, im, lane, i, 0.5f * f[i] * f[(i + 1) & 7]);
    }

    _Float16* pp = g_PSI + (size_t)s * (2 * NAMP);
#pragma unroll
    for (int r = 0; r < 8; ++r) {
        pp[(r << 5) + lane]        = (_Float16)re[r];
        pp[NAMP + (r << 5) + lane] = (_Float16)im[r];
    }
}

// =====================================================================
// Kernel 3: build U = composition of the 4 variational layers.
// =====================================================================
__global__ __launch_bounds__(256) void u_kernel(const float* __restrict__ var_w)
{
    const int lane = threadIdx.x & 31;
    const int wave = threadIdx.x >> 5;
    const int w    = blockIdx.x * 8 + wave;   // 0..255

    float re[8], im[8];
#pragma unroll
    for (int r = 0; r < 8; ++r) {
        re[r] = (r == (w >> 5) && lane == (w & 31)) ? 1.f : 0.f;
        im[r] = 0.f;
    }

    for (int l = 0; l < 4; ++l) {
        const float* vw = var_w + l * (NQ * 3);
#pragma unroll
        for (int i = 0; i < NQ; ++i) {
            g_rx(re, im, lane, i, vw[3 * i + 0]);
            g_ry(re, im, lane, i, vw[3 * i + 1]);
            g_rz(re, im, lane, i, vw[3 * i + 2]);
        }
        if ((l & 1) == 0) {
#pragma unroll
            for (int i = 0; i < NQ - 1; ++i) g_cnot(re, im, lane, i, i + 1);
        } else {
#pragma unroll
            for (int i = 0; i < NQ; ++i) g_cnot(re, im, lane, i, (i + 1) & 7);
        }
#pragma unroll
        for (int i = 0; i < NQ; i += 2) g_cry(re, im, lane, i, i + 1, vw[3 * i + 0]);
    }

#pragma unroll
    for (int r = 0; r < 8; ++r) {
        const int j = (r << 5) | lane;
        g_UR[(size_t)w * NAMP + j] = (_Float16)re[r];
        g_UI[(size_t)w * NAMP + j] = (_Float16)im[r];
    }
}

// =====================================================================
// Kernel 4: repack B_R = [ReU^T; -ImU^T], B_I = [ImU^T; ReU^T] into
// per-lane WMMA fragment order -> contiguous 32B loads in the GEMM.
// =====================================================================
__global__ __launch_bounds__(256) void repack_kernel()
{
    const int tid  = blockIdx.x * 256 + threadIdx.x;   // 0..131071
    const int jj   = tid & 15;
    const int lane = (tid >> 4) & 31;
    const int nt   = (tid >> 9) & 15;
    const int kc   = (tid >> 13) & 15;

    const int k = kc * 32 + ((lane >> 4) * 8) + (jj < 8 ? jj : jj + 8);
    const int n = nt * 16 + (lane & 15);

    _Float16 vR, vI;
    if (k < NAMP) {
        vR = g_UR[(size_t)k * NAMP + n];
        vI = g_UI[(size_t)k * NAMP + n];
    } else {
        const int k2 = k - NAMP;
        vR = (_Float16)(-(float)g_UI[(size_t)k2 * NAMP + n]);
        vI = g_UR[(size_t)k2 * NAMP + n];
    }
    g_BWR[tid] = vR;
    g_BWI[tid] = vI;
}

// =====================================================================
// Kernel 5: psi_out = U*psi (complex GEMM M=8192 K=512 N=256, WMMA) with
// async double-buffered B staging in LDS, then z = probs@signs (WMMA).
// One wave = 16 samples; 4 waves / block. B slices shared block-wide.
// =====================================================================
__device__ __forceinline__ void issue_b_slice(_Float16* buf, int nt, int tid)
{
#pragma unroll
    for (int i = 0; i < 16; ++i) {
        const int idx   = tid + 128 * i;          // 0..2047 chunks of 16B
        const int b     = idx & 1;                // low/high 16B of a fragment
        const int lane2 = (idx >> 1) & 31;
        const int kc    = (idx >> 6) & 15;
        const int m     = (idx >> 10) & 1;        // 0 = B_R, 1 = B_I
        const _Float16* gsrc = (m ? g_BWI : g_BWR)
                             + (size_t)(((kc * 16 + nt) * 32 + lane2) * 16 + b * 8);
        _Float16* ldst = buf + (((kc * 2 + m) * 32 + lane2) * 16 + b * 8);
        copy16_async(gsrc, ldst);
    }
}

__global__ __launch_bounds__(128) void gemm_kernel()
{
    __shared__ _Float16 ldsA[4][16 * 512];             // psi rows, 16KB/wave
    __shared__ _Float16 ldsP[4][16 * 256];             // prob tile, 8KB/wave
    __shared__ _Float16 ldsB[2][16 * 2 * 32 * 16];     // B slice double buffer, 2x32KB

    const int tid   = threadIdx.x;
    const int lane  = tid & 31;
    const int wave  = tid >> 5;              // 0..3
    const int row0  = (blockIdx.x * 4 + wave) * 16;
    const int mrow  = lane & 15;
    const int kbase = (lane >> 4) * 8;
    const int m0    = (lane >> 4) * 8;

    // stage this wave's 16 psi rows into LDS (async, 16B chunks)
    {
        const _Float16* src = g_PSI + (size_t)row0 * 512;
        _Float16* dst = ldsA[wave];
        for (int t = lane; t < 1024; t += 32)
            copy16_async(src + t * 8, dst + t * 8);
    }
    // prologue: B slice for nt = 0
    issue_b_slice(ldsB[0], 0, tid);

    for (int nt = 0; nt < 16; ++nt) {
        wait_async_barrier();                        // asynccnt==0 + block barrier
        if (nt < 15) issue_b_slice(ldsB[(nt + 1) & 1], nt + 1, tid);
        const _Float16* bb = ldsB[nt & 1];

        v8f accR = {}, accI = {};
        for (int kc = 0; kc < 16; ++kc) {
            v16h A, BR, BI;
            const _Float16* ap = &ldsA[wave][mrow * 512 + kc * 32 + kbase];
#pragma unroll
            for (int j = 0; j < 8; ++j) {
                A[j]     = ap[j];
                A[8 + j] = ap[16 + j];
            }
            const _Float16* bpR = bb + (((kc * 2 + 0) * 32 + lane) * 16);
            const _Float16* bpI = bb + (((kc * 2 + 1) * 32 + lane) * 16);
#pragma unroll
            for (int j = 0; j < 16; ++j) { BR[j] = bpR[j]; BI[j] = bpI[j]; }
            accR = __builtin_amdgcn_wmma_f32_16x16x32_f16(
                       false, A, false, BR, (short)0, accR, false, false);
            accI = __builtin_amdgcn_wmma_f32_16x16x32_f16(
                       false, A, false, BI, (short)0, accI, false, false);
        }
#pragma unroll
        for (int r = 0; r < 8; ++r) {
            float p = accR[r] * accR[r] + accI[r] * accI[r];
            ldsP[wave][(m0 + r) * 256 + nt * 16 + mrow] = (_Float16)p;
        }
    }
    __syncthreads();

    // ---- z = probs @ signs via WMMA (signs generated in VALU) ----
    v8f accZ = {};
    for (int kc = 0; kc < 8; ++kc) {
        v16h A3, B3;
        const _Float16* hp = &ldsP[wave][mrow * 256 + kc * 32 + kbase];
#pragma unroll
        for (int j = 0; j < 8; ++j) {
            A3[j]     = hp[j];
            A3[8 + j] = hp[16 + j];
        }
        const bool valid = mrow < NQ;
        const int  shift = 7 - (mrow & 7);
#pragma unroll
        for (int j = 0; j < 16; ++j) {
            const int k = kc * 32 + kbase + (j < 8 ? j : j + 8);
            float sv = ((k >> shift) & 1) ? -1.f : 1.f;
            B3[j] = valid ? (_Float16)sv : (_Float16)0.f;
        }
        accZ = __builtin_amdgcn_wmma_f32_16x16x32_f16(
                   false, A3, false, B3, (short)0, accZ, false, false);
    }
    if (mrow < NQ) {
#pragma unroll
        for (int r = 0; r < 8; ++r)
            g_Z[(size_t)(row0 + m0 + r) * NQ + mrow] = accZ[r];
    }
}

// =====================================================================
// Kernel 6: classifier. One thread per sample (weights -> scalar loads).
// =====================================================================
__global__ __launch_bounds__(256) void cls_kernel(
    const float* __restrict__ cw1, const float* __restrict__ cb1,
    const float* __restrict__ cw2, const float* __restrict__ cb2,
    const float* __restrict__ cw3, const float* __restrict__ cb3,
    float* __restrict__ out)
{
    const int s = blockIdx.x * 256 + threadIdx.x;   // 0..8191

    float z[NQ];
#pragma unroll
    for (int i = 0; i < NQ; ++i) z[i] = g_Z[(size_t)s * NQ + i];

    float h2[16];
#pragma unroll
    for (int t = 0; t < 16; ++t) h2[t] = cb2[t];

    for (int j = 0; j < 32; ++j) {
        float h1 = cb1[j];
#pragma unroll
        for (int k = 0; k < NQ; ++k) h1 = fmaf(z[k], cw1[j * NQ + k], h1);
        h1 = fmaxf(h1, 0.f);
#pragma unroll
        for (int t = 0; t < 16; ++t) h2[t] = fmaf(h1, cw2[t * 32 + j], h2[t]);
    }

    float o[OUT_DIM];
#pragma unroll
    for (int u = 0; u < OUT_DIM; ++u) o[u] = cb3[u];
#pragma unroll
    for (int t = 0; t < 16; ++t) {
        float ht = fmaxf(h2[t], 0.f);
#pragma unroll
        for (int u = 0; u < OUT_DIM; ++u) o[u] = fmaf(ht, cw3[u * 16 + t], o[u]);
    }
#pragma unroll
    for (int u = 0; u < OUT_DIM; ++u) out[(size_t)s * OUT_DIM + u] = o[u];
}

// =====================================================================
extern "C" void kernel_launch(void* const* d_in, const int* in_sizes, int n_in,
                              void* d_out, int out_size, void* d_ws, size_t ws_size,
                              hipStream_t stream)
{
    (void)in_sizes; (void)n_in; (void)d_ws; (void)ws_size; (void)out_size;

    const float* x     = (const float*)d_in[0];
    const float* w1    = (const float*)d_in[1];
    const float* b1    = (const float*)d_in[2];
    const float* w2    = (const float*)d_in[3];
    const float* b2    = (const float*)d_in[4];
    const float* fm_w  = (const float*)d_in[5];
    const float* var_w = (const float*)d_in[6];
    const float* cw1   = (const float*)d_in[7];
    const float* cb1   = (const float*)d_in[8];
    const float* cw2   = (const float*)d_in[9];
    const float* cb2   = (const float*)d_in[10];
    const float* cw3   = (const float*)d_in[11];
    const float* cb3   = (const float*)d_in[12];
    float* out = (float*)d_out;

    hipLaunchKernelGGL(feats_kernel,  dim3(BATCH / 128), dim3(256), 0, stream,
                       x, w1, b1, w2, b2);
    hipLaunchKernelGGL(u_kernel,      dim3(NAMP / 8),    dim3(256), 0, stream, var_w);
    hipLaunchKernelGGL(repack_kernel, dim3(512),         dim3(256), 0, stream);
    hipLaunchKernelGGL(fm_kernel,     dim3(BATCH / 8),   dim3(256), 0, stream, fm_w);
    hipLaunchKernelGGL(gemm_kernel,   dim3(BATCH / 64),  dim3(128), 0, stream);
    hipLaunchKernelGGL(cls_kernel,    dim3(BATCH / 256), dim3(256), 0, stream,
                       cw1, cb1, cw2, cb2, cw3, cb3, out);
}
